// ParticleKernel_30434138259762
// MI455X (gfx1250) — matrix operand order repaired and checked by reading, not compiled, not run on hardware
//
#include <hip/hip_runtime.h>
#include <hip/hip_bf16.h>

// Problem constants (match reference)
#define BB 4
#define NN 2048
#define DD 512
#define PP 16
#define KK 4
#define KD 2048        // K*D

typedef __attribute__((ext_vector_type(16))) _Float16 v16h;
typedef __attribute__((ext_vector_type(8)))  _Float16 v8h;
typedef __attribute__((ext_vector_type(8)))  float    v8f;

#define LOG2E 1.44269504088896340736f

// ---------------------------------------------------------------------------
// WMMA helper: D = A(16x32 f16) * B(32x16 f16) + C(16x16 f32)
// ---------------------------------------------------------------------------
__device__ __forceinline__ v8f wmma_f16(v16h a, v16h b, v8f c) {
  return __builtin_amdgcn_wmma_f32_16x16x32_f16(
      /*neg_a=*/false, a, /*neg_b=*/false, b,
      /*c_mod=*/(short)0, c, /*reuse_a=*/false, /*reuse_b=*/false);
}

// Load a 16x32 f16 A-operand slice for this lane from a row-major f16 matrix.
// ISA 7.12.2 (16-bit A 16x32): lane L holds row m=L%16; with g=L/16,
//   vector elems 0..7  <-> K = g*8 + 0..7
//   vector elems 8..15 <-> K = 16 + g*8 + 0..7
__device__ __forceinline__ v16h load_A_f16(const _Float16* rowptr, int g) {
  union { v16h v; v8h h[2]; } u;
  u.h[0] = *(const v8h*)(rowptr + g * 8);
  u.h[1] = *(const v8h*)(rowptr + 16 + g * 8);
  return u.v;
}

// ---------------------------------------------------------------------------
// Prep kernels
// ---------------------------------------------------------------------------
__global__ void cvt_f16_kernel(const float* __restrict__ in,
                               _Float16* __restrict__ out, int n) {
  int i = blockIdx.x * 256 + threadIdx.x;
  if (i < n) out[i] = (_Float16)in[i];
}

// in (rows, cols) f32 -> out (cols, rows) f16
__global__ void transpose_f16_kernel(const float* __restrict__ in,
                                     _Float16* __restrict__ out,
                                     int rows, int cols) {
  int i = blockIdx.x * 256 + threadIdx.x;
  if (i < rows * cols) {
    int r = i / cols, c = i % cols;
    out[(size_t)c * rows + r] = (_Float16)in[i];
  }
}

// position (B,N,P) f32 -> posT (B,P,N) f16  and  qn[b,j] = |pos_j|^2
__global__ void posprep_kernel(const float* __restrict__ pos,
                               _Float16* __restrict__ posT,
                               float* __restrict__ qn) {
  int i = blockIdx.x * 256 + threadIdx.x;           // over B*N
  if (i < BB * NN) {
    int b = i / NN, j = i % NN;
    const float* p = pos + (size_t)i * PP;
    float q = 0.f;
    #pragma unroll
    for (int t = 0; t < PP; ++t) {
      float x = p[t];
      q += x * x;
      posT[((size_t)b * PP + t) * NN + j] = (_Float16)x;
    }
    qn[i] = q;
  }
}

// ---------------------------------------------------------------------------
// Kernel A: proj[b,j,k*D+d] = mass[b,j] * sum_e charge[b,j,e] * W_charge[kD+d,e]
// ---------------------------------------------------------------------------
__global__ __launch_bounds__(256) void proj_kernel(
    const _Float16* __restrict__ chg,    // (B,N,D) f16
    const _Float16* __restrict__ WT,     // (D, K*D) f16 = W_charge^T
    const float* __restrict__ mass,      // (B,N)
    _Float16* __restrict__ proj) {       // (B,N,K*D)
  const int NB16 = NN / 16;
  int blk = blockIdx.x;
  int b = blk / NB16;
  int i0 = (blk % NB16) * 16;
  int tid = threadIdx.x, wave = tid >> 5, lane = tid & 31;
  int mrow = lane & 15, g = lane >> 4;
  int c0 = wave * 256;                    // 16 tiles of 16 columns

  v8f acc[16] = {};
  const _Float16* arow = chg + (size_t)(b * NN + i0 + mrow) * DD;

  for (int e0 = 0; e0 < DD; e0 += 32) {
    v16h av = load_A_f16(arow + e0, g);
    const _Float16* bb = WT + (size_t)(e0 + lane) * KD + c0;  // lane = K row
    __builtin_prefetch(bb + (size_t)32 * KD, 0, 3);           // next e-step
    #pragma unroll
    for (int t = 0; t < 16; ++t) {
      v16h bv;
      #pragma unroll
      for (int s = 0; s < 16; ++s) bv[s] = WT[(size_t)(e0 + ((s < 8) ? g * 8 + s : 16 + g * 8 + (s - 8))) * KD + c0 + t * 16 + mrow];
      acc[t] = wmma_f16(av, bv, acc[t]);
    }
  }
  // row scale by mass, store f16.  C layout: row = r + g*8, col = lane&15.
  float mv[8];
  #pragma unroll
  for (int r = 0; r < 8; ++r) mv[r] = mass[b * NN + i0 + g * 8 + r];
  #pragma unroll
  for (int t = 0; t < 16; ++t) {
    #pragma unroll
    for (int r = 0; r < 8; ++r) {
      size_t off = (size_t)(b * NN + i0 + g * 8 + r) * KD + c0 + t * 16 + mrow;
      proj[off] = (_Float16)(acc[t][r] * mv[r]);
    }
  }
}

// ---------------------------------------------------------------------------
// Kernel B (main): field[b,i,k*D+d] = sum_j exp(-c_k*dist2(i,j)) * proj[b,j,k,d]
// One block per (b, 16-row i-tile). 4 waves: wave = k, each wave owns all 512
// d-columns of its k (32 f32 accumulator tiles = 256 VGPRs), so the weight
// prep (2 Gram WMMAs + 16 exp2 + LDS round trip) amortizes over 32 WMMAs.
// ---------------------------------------------------------------------------
__global__ __launch_bounds__(128) void field_kernel(
    const float* __restrict__ pos,       // (B,N,P) f32
    const _Float16* __restrict__ posT,   // (B,P,N) f16
    const float* __restrict__ qn,        // (B,N) f32
    const float* __restrict__ log_bw,    // (K)
    const _Float16* __restrict__ proj,   // (B,N,K*D) f16 (mass folded in)
    _Float16* __restrict__ field) {      // (B,N,K*D) f16
  const int NB16 = NN / 16;
  int blk = blockIdx.x;
  int b = blk / NB16;
  int i0 = (blk % NB16) * 16;
  int tid = threadIdx.x, wave = tid >> 5, lane = tid & 31;  // wave = k (0..3)
  int mrow = lane & 15, g = lane >> 4;
  int k = wave;
  int colbase = k * DD;                  // this wave's 512-column slice of K*D

  __shared__ float g2[4][16][34];        // per-wave 16x32 Gram staging (+pad)

  // kernel scale: exp(-c*x) = exp2(s*x), c = 1/(2*exp(lb_k)+eps)
  float bw = __builtin_amdgcn_exp2f(log_bw[k] * LOG2E);
  float c  = 1.0f / (2.0f * bw + 1e-8f);
  float s  = -c * LOG2E;

  // A-side positions (constant over j loop): elems 0..7 = dims g*8..g*8+7,
  // elems 8..15 = 0 (K padded 16 -> 32).
  union { v16h v; _Float16 e[16]; } pa;
  {
    const float* prow = pos + (size_t)(b * NN + i0 + mrow) * PP + g * 8;
    #pragma unroll
    for (int h = 0; h < 8; ++h) pa.e[h] = (_Float16)prow[h];
    #pragma unroll
    for (int h = 8; h < 16; ++h) pa.e[h] = (_Float16)0.0f;
  }
  float a_i = s * qn[b * NN + i0 + mrow];

  v8f acc[32] = {};
  const _Float16* projB = proj + (size_t)b * NN * KD;
  const _Float16* posTb = posT + (size_t)b * PP * NN;
  const float*    qnb   = qn + b * NN;

  for (int j0 = 0; j0 < NN; j0 += 32) {
    // --- Gram tiles via f16 WMMA (B rows 16..31 are zero padding) ---
    #pragma unroll
    for (int t2 = 0; t2 < 2; ++t2) {
      v16h pb = {};
      #pragma unroll
      for (int s = 0; s < 8; ++s) pb[s] = posTb[(size_t)(g * 8 + s) * NN + j0 + t2 * 16 + mrow];
      v8f zc = {};
      v8f gz = wmma_f16(pa.v, pb, zc);
      // C layout -> LDS: row r+g*8, col t2*16 + (lane&15)
      #pragma unroll
      for (int r = 0; r < 8; ++r)
        g2[wave][r + g * 8][t2 * 16 + mrow] = gz[r];
    }
    __builtin_amdgcn_wave_barrier();      // same-wave LDS is processed in-order

    // --- re-read G in A-operand pattern, apply exp2, pack f16 weights ---
    union { v16h v; _Float16 e[16]; } wa;
    const float* grow = &g2[wave][mrow][0];
    #pragma unroll
    for (int h = 0; h < 16; ++h) {
      int jr = (h < 8) ? (g * 8 + h) : (16 + g * 8 + (h - 8));
      float gv  = grow[jr];
      float arg = a_i + s * qnb[j0 + jr] - 2.0f * s * gv;
      wa.e[h] = (_Float16)__builtin_amdgcn_exp2f(arg);
    }
    __builtin_amdgcn_wave_barrier();

    // --- main GEMM: 32 WMMAs over this wave's 512 columns, with a 4-deep
    // rotating B-tile buffer to keep loads for 4 tiles in flight ---
    const _Float16* bb = projB + (size_t)(j0 + lane) * KD + colbase;
    __builtin_prefetch(bb + (size_t)32 * KD, 0, 3);   // next j-step's B rows
    v16h bvbuf[4];
    #pragma unroll
    for (int t = 0; t < 4; ++t)
      for (int s = 0; s < 16; ++s) bvbuf[t][s] = projB[(size_t)(j0 + ((s < 8) ? g * 8 + s : 16 + g * 8 + (s - 8))) * KD + colbase + t * 16 + mrow];
    #pragma unroll
    for (int t = 0; t < 32; ++t) {
      v16h cur = bvbuf[t & 3];
      if (t + 4 < 32)
        for (int s = 0; s < 16; ++s) bvbuf[t & 3][s] = projB[(size_t)(j0 + ((s < 8) ? g * 8 + s : 16 + g * 8 + (s - 8))) * KD + colbase + (t + 4) * 16 + mrow];
      acc[t] = wmma_f16(wa.v, cur, acc[t]);
    }
  }

  // epilogue: C layout row = r+g*8, col = lane&15
  #pragma unroll
  for (int t = 0; t < 32; ++t) {
    #pragma unroll
    for (int r = 0; r < 8; ++r) {
      size_t off = (size_t)(b * NN + i0 + g * 8 + r) * KD + colbase + t * 16 + mrow;
      field[off] = (_Float16)acc[t][r];
    }
  }
}

// ---------------------------------------------------------------------------
// Kernel C: out = LayerNorm(field @ W_combine^T) * gamma + beta
// ---------------------------------------------------------------------------
__global__ __launch_bounds__(256) void combine_ln_kernel(
    const _Float16* __restrict__ field,  // (B,N,KD) f16
    const _Float16* __restrict__ WT,     // (KD, D) f16 = W_combine^T
    const float* __restrict__ gamma,
    const float* __restrict__ beta,
    float* __restrict__ out) {           // (B,N,D) f32
  const int NB16 = NN / 16;
  int blk = blockIdx.x;
  int b = blk / NB16;
  int i0 = (blk % NB16) * 16;
  int tid = threadIdx.x, wave = tid >> 5, lane = tid & 31;
  int nn = lane & 15, g = lane >> 4;
  int d0 = wave * 64;                     // 4 tiles of 16 columns

  __shared__ float fld[16][520];
  __shared__ float red1[16][16], red2[16][16];
  __shared__ float mu[16], rs[16];

  v8f acc[4] = {};
  const _Float16* arow = field + (size_t)(b * NN + i0 + nn) * KD;

  for (int e0 = 0; e0 < KD; e0 += 32) {
    v16h av = load_A_f16(arow + e0, g);
    const _Float16* bb = WT + (size_t)(e0 + lane) * DD + d0;
    __builtin_prefetch(bb + (size_t)32 * DD, 0, 3);   // next e-step
    #pragma unroll
    for (int t = 0; t < 4; ++t) {
      v16h bv;
      #pragma unroll
      for (int s = 0; s < 16; ++s) bv[s] = WT[(size_t)(e0 + ((s < 8) ? g * 8 + s : 16 + g * 8 + (s - 8))) * DD + d0 + t * 16 + nn];
      acc[t] = wmma_f16(av, bv, acc[t]);
    }
  }
  #pragma unroll
  for (int t = 0; t < 4; ++t)
    #pragma unroll
    for (int r = 0; r < 8; ++r)
      fld[r + g * 8][d0 + t * 16 + nn] = acc[t][r];
  __syncthreads();

  // LayerNorm over D=512 per row; 16 threads per row, 32 cols each.
  int row = tid >> 4, sub = tid & 15;
  float s1 = 0.f, s2 = 0.f;
  for (int cix = sub * 32; cix < sub * 32 + 32; ++cix) {
    float x = fld[row][cix];
    s1 += x; s2 += x * x;
  }
  red1[row][sub] = s1; red2[row][sub] = s2;
  __syncthreads();
  if (sub == 0) {
    float t1 = 0.f, t2 = 0.f;
    for (int q = 0; q < 16; ++q) { t1 += red1[row][q]; t2 += red2[row][q]; }
    float mean = t1 * (1.0f / DD);
    float var  = t2 * (1.0f / DD) - mean * mean;
    mu[row] = mean;
    rs[row] = rsqrtf(var + 1e-5f);
  }
  __syncthreads();
  float mean = mu[row], rstd = rs[row];
  size_t orow = (size_t)(b * NN + i0 + row) * DD;
  for (int cix = sub * 32; cix < sub * 32 + 32; ++cix) {
    out[orow + cix] = (fld[row][cix] - mean) * rstd * gamma[cix] + beta[cix];
  }
}

// ---------------------------------------------------------------------------
// Host launcher
// ---------------------------------------------------------------------------
extern "C" void kernel_launch(void* const* d_in, const int* in_sizes, int n_in,
                              void* d_out, int out_size, void* d_ws, size_t ws_size,
                              hipStream_t stream) {
  const float* charge   = (const float*)d_in[0];   // (B,N,D)
  const float* position = (const float*)d_in[1];   // (B,N,P)
  const float* mass     = (const float*)d_in[2];   // (B,N,1)
  const float* log_bw   = (const float*)d_in[3];   // (K)
  const float* W_charge = (const float*)d_in[4];   // (K*D, D)
  const float* W_comb   = (const float*)d_in[5];   // (D, K*D)
  const float* gamma    = (const float*)d_in[6];   // (D)
  const float* beta     = (const float*)d_in[7];   // (D)
  float* out = (float*)d_out;                      // (B,N,D)

  // Workspace carve-out (all 256B aligned). Total ~80 MB.
  size_t off = 0;
  auto carve = [&](size_t bytes) -> void* {
    void* p = (char*)d_ws + off;
    off += (bytes + 255) & ~(size_t)255;
    return p;
  };
  _Float16* chg_h = (_Float16*)carve((size_t)BB * NN * DD * 2);
  _Float16* WcgT  = (_Float16*)carve((size_t)DD * KD * 2);
  _Float16* WcbT  = (_Float16*)carve((size_t)KD * DD * 2);
  _Float16* posT  = (_Float16*)carve((size_t)BB * PP * NN * 2);
  float*    qn    = (float*)   carve((size_t)BB * NN * 4);
  _Float16* proj  = (_Float16*)carve((size_t)BB * NN * KD * 2);
  _Float16* fld   = (_Float16*)carve((size_t)BB * NN * KD * 2);

  const int nchg = BB * NN * DD;
  cvt_f16_kernel<<<(nchg + 255) / 256, 256, 0, stream>>>(charge, chg_h, nchg);
  transpose_f16_kernel<<<(KD * DD + 255) / 256, 256, 0, stream>>>(
      W_charge, WcgT, KD, DD);                     // (2048,512) -> (512,2048)
  transpose_f16_kernel<<<(DD * KD + 255) / 256, 256, 0, stream>>>(
      W_comb, WcbT, DD, KD);                       // (512,2048) -> (2048,512)
  posprep_kernel<<<(BB * NN + 255) / 256, 256, 0, stream>>>(position, posT, qn);

  const int nblk = BB * (NN / 16);                 // 512
  proj_kernel<<<nblk, 256, 0, stream>>>(chg_h, WcgT, mass, proj);
  field_kernel<<<nblk, 128, 0, stream>>>(position, posT, qn, log_bw, proj, fld);
  combine_ln_kernel<<<nblk, 256, 0, stream>>>(fld, WcbT, gamma, beta, out);
}
